// VGNConvLayer_51075751084772
// MI455X (gfx1250) — matrix operand order repaired and checked by reading, not compiled, not run on hardware
//
#include <hip/hip_runtime.h>

// ---------------- problem constants (fixed by the reference) ----------------
constexpr int   N_NODES = 50000;
constexpr int   N_EDGES = 800000;
constexpr int   DIM     = 256;
constexpr int   NC      = 4;
constexpr float BN_EPS  = 1e-5f;

typedef __attribute__((ext_vector_type(16))) __bf16 v16bf;
typedef __attribute__((ext_vector_type(8)))  float  v8f;
typedef __attribute__((ext_vector_type(4)))  float  nf4;   // native vec for NT loads

union FragBF {
    v16bf bf;
    uint4 q[2];
};

__device__ __forceinline__ unsigned short f2bf(float f) {
    unsigned int u = __float_as_uint(f);
    unsigned int r = u + 0x7FFFu + ((u >> 16) & 1u);   // round-to-nearest-even
    return (unsigned short)(r >> 16);
}

// ---------------- weight convert + transpose (once) -------------------------
__global__ void convert_weights_kernel(const float* __restrict__ W1,
                                       const float* __restrict__ W2,
                                       unsigned short* __restrict__ W1T,
                                       unsigned short* __restrict__ W2T) {
    int tid = blockIdx.x * blockDim.x + threadIdx.x;      // NC*DIM*DIM
    if (tid >= NC * DIM * DIM) return;
    int c   = tid / (DIM * DIM);
    int rem = tid - c * (DIM * DIM);
    int i   = rem / DIM;           // input dim (row of W)
    int j   = rem - i * DIM;       // output dim (col of W)
    int t   = c * (DIM * DIM) + j * DIM + i;  // transposed: [out][in] contiguous in K
    W1T[t] = f2bf(W1[tid]);
    W2T[t] = f2bf(W2[tid]);
}

// ---------------- simple elementwise kernels --------------------------------
__global__ void copy_f4_kernel(const float4* __restrict__ s, float4* __restrict__ d, int n4) {
    int t = blockIdx.x * blockDim.x + threadIdx.x;
    if (t < n4) d[t] = s[t];
}

__global__ void zero_f4_kernel(float4* __restrict__ d, int n4) {
    int t = blockIdx.x * blockDim.x + threadIdx.x;
    if (t < n4) d[t] = make_float4(0.f, 0.f, 0.f, 0.f);
}

// ---------------- edge message + scatter-add aggregate ----------------------
// 64 threads per edge (4 floats each): coalesced edge_attr stream (non-temporal
// so the 819MB stream does not evict L2-resident x/aggr), coalesced x[src]
// gather, fp32 atomics into L2-resident aggr.
__global__ void edge_msg_kernel(const float* __restrict__ x,
                                const int*   __restrict__ ei,
                                const float* __restrict__ ea,
                                float*       __restrict__ aggr) {
    int gid = blockIdx.x * blockDim.x + threadIdx.x;
    if (gid >= N_EDGES * 64) return;
    int e  = gid >> 6;
    int d4 = (gid & 63) << 2;
    int src = ei[e];
    int dst = ei[N_EDGES + e];
    const float4 xv = *(const float4*)(x + (size_t)src * DIM + d4);
    const nf4 ev =
        __builtin_nontemporal_load((const nf4*)(ea + (size_t)e * DIM + d4));
    float4 r;
    r.x = fmaxf(xv.x + ev.x, 0.f);
    r.y = fmaxf(xv.y + ev.y, 0.f);
    r.z = fmaxf(xv.z + ev.z, 0.f);
    r.w = fmaxf(xv.w + ev.w, 0.f);
    float* o = aggr + (size_t)dst * DIM + d4;
    unsafeAtomicAdd(o + 0, r.x);
    unsafeAtomicAdd(o + 1, r.y);
    unsafeAtomicAdd(o + 2, r.z);
    unsafeAtomicAdd(o + 3, r.w);
}

// ---------------- h0 = bf16((1+eps)*x + aggr) -------------------------------
__global__ void build_h0_kernel(const float4* __restrict__ x,
                                const float4* __restrict__ aggr,
                                const float*  __restrict__ epsv, int c,
                                uint2* __restrict__ h0, int n4) {
    int t = blockIdx.x * blockDim.x + threadIdx.x;
    if (t >= n4) return;
    float s = 1.0f + epsv[c];
    float4 xv = x[t], av = aggr[t];
    unsigned int lo = (unsigned int)f2bf(s * xv.x + av.x) |
                      ((unsigned int)f2bf(s * xv.y + av.y) << 16);
    unsigned int hi = (unsigned int)f2bf(s * xv.z + av.z) |
                      ((unsigned int)f2bf(s * xv.w + av.w) << 16);
    h0[t] = make_uint2(lo, hi);
}

// ---------------- WMMA GEMM: [N,256] x [256,256] ----------------------------
// Wave tile 16(M) x 128(N); block = 8 waves -> 64(M) x 256(N).
// MODE 0: out_bf16 = relu(A*W + b)
// MODE 1: out_f32  = mask[m]*(A*W + b) + xres ; fused BN sum/sumsq atomics
template <int MODE>
__global__ void gemm_wmma_kernel(const unsigned short* __restrict__ A,
                                 const unsigned short* __restrict__ WT,
                                 const float* __restrict__ bias,
                                 const float* __restrict__ mask,
                                 const float* __restrict__ xres,
                                 unsigned short* __restrict__ outB,
                                 float*          __restrict__ outF,
                                 float*          __restrict__ stats,
                                 int nRows) {
    const int lane  = threadIdx.x & 31;
    const int wave  = threadIdx.x >> 5;
    const int mBase = blockIdx.x * 64 + (wave >> 1) * 16;
    const int nBase = (wave & 1) * 128;
    const int half  = lane >> 4;      // 0 or 1
    const int l15   = lane & 15;

    v8f zero = {};
    v8f acc[8];
#pragma unroll
    for (int j = 0; j < 8; ++j) acc[j] = zero;

    int rowA = mBase + l15;
    if (rowA >= nRows) rowA = nRows - 1;           // clamp: keep EXEC all-ones
    const unsigned short* aRow = A + (size_t)rowA * DIM + half * 8;
    const unsigned short* bBase = WT + (size_t)(nBase + l15) * DIM + half * 8;

    for (int k0 = 0; k0 < DIM; k0 += 32) {
        FragBF a;
        a.q[0] = *(const uint4*)(aRow + k0);        // K = k0+{0..7}|{8..15}
        a.q[1] = *(const uint4*)(aRow + k0 + 16);   // K = k0+{16..23}|{24..31}
#pragma unroll
        for (int j = 0; j < 8; j += 2) {            // paired tiles: 4 loads in flight
            const unsigned short* bc0 = bBase + (size_t)(j * 16) * DIM + k0;
            const unsigned short* bc1 = bBase + (size_t)((j + 1) * 16) * DIM + k0;
            FragBF b0, b1;
            b0.q[0] = *(const uint4*)(bc0);
            b0.q[1] = *(const uint4*)(bc0 + 16);
            b1.q[0] = *(const uint4*)(bc1);
            b1.q[1] = *(const uint4*)(bc1 + 16);
            acc[j] = __builtin_amdgcn_wmma_f32_16x16x32_bf16(
                false, a.bf, false, b0.bf, (short)0, acc[j], false, false);
            acc[j + 1] = __builtin_amdgcn_wmma_f32_16x16x32_bf16(
                false, a.bf, false, b1.bf, (short)0, acc[j + 1], false, false);
        }
    }

    // epilogue: lane holds column n_local = l15; VGPR r -> row m = r + 8*half
    float mval[8];
    int   mrow[8];
#pragma unroll
    for (int r = 0; r < 8; ++r) {
        int m = mBase + r + half * 8;
        mrow[r] = m;
        mval[r] = (MODE == 1 && m < nRows) ? mask[m] : 0.f;
    }
#pragma unroll
    for (int j = 0; j < 8; ++j) {
        int n = nBase + j * 16 + l15;
        float bn = bias[n];
        float cs = 0.f, cq = 0.f;
#pragma unroll
        for (int r = 0; r < 8; ++r) {
            int m = mrow[r];
            if (m < nRows) {
                float v = acc[j][r] + bn;
                if (MODE == 0) {
                    v = fmaxf(v, 0.f);
                    outB[(size_t)m * DIM + n] = f2bf(v);
                } else {
                    float xp = mval[r] * v + xres[(size_t)m * DIM + n];
                    outF[(size_t)m * DIM + n] = xp;
                    cs += xp;
                    cq += xp * xp;
                }
            }
        }
        if (MODE == 1) {                 // fused BN partials (stats pre-zeroed)
            unsafeAtomicAdd(&stats[n], cs);
            unsafeAtomicAdd(&stats[DIM + n], cq);
        }
    }
}

__device__ __forceinline__ float bn1(float v, int d,
                                     const float* __restrict__ stats,
                                     const float* __restrict__ gamma,
                                     const float* __restrict__ beta) {
    constexpr float invN = 1.0f / (float)N_NODES;
    float mean = stats[d] * invN;
    float var  = stats[DIM + d] * invN - mean * mean;
    float inv  = rsqrtf(var + BN_EPS);
    return gamma[d] * (v - mean) * inv + beta[d];
}

__global__ void bn_norm_kernel(const float4* __restrict__ xn,
                               const float*  __restrict__ stats,
                               const float*  __restrict__ gamma,
                               const float*  __restrict__ beta,
                               float4* __restrict__ xo, int n4) {
    int t = blockIdx.x * blockDim.x + threadIdx.x;
    if (t >= n4) return;
    int d0 = (t & (DIM / 4 - 1)) * 4;
    float4 v = xn[t];
    float4 o;
    o.x = bn1(v.x, d0 + 0, stats, gamma, beta);
    o.y = bn1(v.y, d0 + 1, stats, gamma, beta);
    o.z = bn1(v.z, d0 + 2, stats, gamma, beta);
    o.w = bn1(v.w, d0 + 3, stats, gamma, beta);
    xo[t] = o;
}

// ---------------- final: out = x_in + relu(x) -------------------------------
__global__ void final_kernel(const float* __restrict__ xin,
                             const float4* __restrict__ xc,
                             float4* __restrict__ out, int n4) {
    int t = blockIdx.x * blockDim.x + threadIdx.x;
    if (t >= n4) return;
    nf4 a = __builtin_nontemporal_load((const nf4*)(xin) + t);  // read-once stream
    float4 b = xc[t];
    out[t] = make_float4(a.x + fmaxf(b.x, 0.f), a.y + fmaxf(b.y, 0.f),
                         a.z + fmaxf(b.z, 0.f), a.w + fmaxf(b.w, 0.f));
}

// ---------------- host-side orchestration -----------------------------------
extern "C" void kernel_launch(void* const* d_in, const int* in_sizes, int n_in,
                              void* d_out, int out_size, void* d_ws, size_t ws_size,
                              hipStream_t stream) {
    const float* x     = (const float*)d_in[0];
    const int*   ei    = (const int*)  d_in[1];
    const float* ea    = (const float*)d_in[2];
    const float* masks = (const float*)d_in[3];
    // d_in[4] complement_masks unused
    const float* W1    = (const float*)d_in[5];
    const float* b1    = (const float*)d_in[6];
    const float* W2    = (const float*)d_in[7];
    const float* b2    = (const float*)d_in[8];
    const float* epsv  = (const float*)d_in[9];
    const float* gamma = (const float*)d_in[10];
    const float* beta  = (const float*)d_in[11];
    float* out = (float*)d_out;

    const size_t NX = (size_t)N_NODES * DIM;        // 12.8M elems
    const size_t WW = (size_t)NC * DIM * DIM;       // per weight tensor

    float*          buf_x  = (float*)d_ws;                 // current x  [N,D] f32
    float*          buf_nx = buf_x + NX;                   // aggr / pre-BN x [N,D] f32
    unsigned short* h0     = (unsigned short*)(buf_nx + NX);  // GEMM1 input bf16
    unsigned short* h1     = h0 + NX;                         // GEMM2 input bf16
    unsigned short* w1t    = h1 + NX;                         // W1^T bf16
    unsigned short* w2t    = w1t + WW;                        // W2^T bf16
    float*          stats  = (float*)(w2t + WW);              // 512 f32

    const int n4      = (int)(NX / 4);
    const int ewBlk   = (n4 + 255) / 256;
    const int wBlk    = (int)((WW + 255) / 256);
    const int edgeBlk = (N_EDGES * 64 + 255) / 256;
    const int gemmBlk = (N_NODES + 63) / 64;

    convert_weights_kernel<<<wBlk, 256, 0, stream>>>(W1, W2, w1t, w2t);
    copy_f4_kernel<<<ewBlk, 256, 0, stream>>>((const float4*)x, (float4*)buf_x, n4);

    for (int c = 0; c < NC; ++c) {
        // aggr = segment_sum(relu(x[src] + e), dst)
        zero_f4_kernel<<<ewBlk, 256, 0, stream>>>((float4*)buf_nx, n4);
        edge_msg_kernel<<<edgeBlk, 256, 0, stream>>>(buf_x, ei, ea, buf_nx);
        // h0 = bf16((1+eps)*x + aggr)
        build_h0_kernel<<<ewBlk, 256, 0, stream>>>((const float4*)buf_x,
                                                   (const float4*)buf_nx,
                                                   epsv, c, (uint2*)h0, n4);
        // h1 = relu(h0 @ W1 + b1)  (bf16 out)
        gemm_wmma_kernel<0><<<gemmBlk, 256, 0, stream>>>(
            h0, w1t + (size_t)c * DIM * DIM, b1 + c * DIM,
            nullptr, nullptr, h1, nullptr, nullptr, N_NODES);
        // x_pre_bn = mask*(h1 @ W2 + b2) + x  (f32 into buf_nx) + fused BN stats
        zero_f4_kernel<<<1, 128, 0, stream>>>((float4*)stats, 128);
        gemm_wmma_kernel<1><<<gemmBlk, 256, 0, stream>>>(
            h1, w2t + (size_t)c * DIM * DIM, b2 + c * DIM,
            masks + (size_t)c * N_NODES, buf_x, nullptr, buf_nx, stats, N_NODES);
        // batch norm normalize -> buf_x
        bn_norm_kernel<<<ewBlk, 256, 0, stream>>>((const float4*)buf_nx, stats,
                                                  gamma + c * DIM, beta + c * DIM,
                                                  (float4*)buf_x, n4);
    }

    final_kernel<<<ewBlk, 256, 0, stream>>>(x, (const float4*)buf_x,
                                            (float4*)out, n4);
}
